// ActorNetAblation_74543452389448
// MI455X (gfx1250) — matrix operand order repaired and checked by reading, not compiled, not run on hardware
//
#include <hip/hip_runtime.h>

typedef __attribute__((ext_vector_type(16))) __bf16 v16bf;
typedef __attribute__((ext_vector_type(8)))  float  v8f;

static constexpr int N  = 50000;
static constexpr int E  = 160000;
static constexpr int T  = 8192;
static constexpr int D  = 32;
static constexpr int AD = 6;

union AFrag { v16bf v; uint4 q[2]; };

// ---------------- setup kernels ----------------

__global__ void k_node_init(const float* __restrict__ x, const float* __restrict__ w,
                            const float* __restrict__ b, float* __restrict__ h) {
  int i = blockIdx.x * blockDim.x + threadIdx.x;
  if (i >= N * D) return;
  int n = i >> 5, d = i & 31;
  float acc = b[d];
#pragma unroll
  for (int k = 0; k < 3; ++k) acc += x[n * 3 + k] * w[k * D + d];
  h[i] = fmaxf(acc, 0.f);
}

__global__ void k_edge_h1(const float* __restrict__ ea, const float* __restrict__ w,
                          const float* __restrict__ b, __bf16* __restrict__ h1) {
  int i = blockIdx.x * blockDim.x + threadIdx.x;
  if (i >= E * D) return;
  int e = i >> 5, d = i & 31;
  float acc = b[d];
#pragma unroll
  for (int k = 0; k < 7; ++k) acc += ea[e * 7 + k] * w[k * D + d];
  h1[i] = (__bf16)fmaxf(acc, 0.f);
}

// pack nn2_w [32,1024] into B fragments: [ct(64)][lane(32)][half(16)]
__global__ void k_pack_nn2(const float* __restrict__ w, __bf16* __restrict__ pb) {
  int i = blockIdx.x * blockDim.x + threadIdx.x;
  if (i >= 64 * 32 * 16) return;
  int j = i & 15, L = (i >> 4) & 31, ct = i >> 9;
  int n = ct * 16 + (L & 15);
  int k = (L >> 4) * 8 + (j & 7) + (j >> 3) * 16;
  pb[i] = (__bf16)w[k * 1024 + n];
}

// pack lin1_w [192,32] into B fragments: [kt(6)][nt(2)][lane(32)][half(16)]
__global__ void k_pack_lin1(const float* __restrict__ w, __bf16* __restrict__ pb) {
  int i = blockIdx.x * blockDim.x + threadIdx.x;
  if (i >= 6 * 2 * 32 * 16) return;
  int j = i & 15, L = (i >> 4) & 31, nt = (i >> 9) & 1, kt = i >> 10;
  int n = nt * 16 + (L & 15);
  int k = kt * 32 + (L >> 4) * 8 + (j & 7) + (j >> 3) * 16;
  pb[i] = (__bf16)w[k * D + n];
}

__global__ void k_deg(const int* __restrict__ dst, float* __restrict__ deg) {
  int e = blockIdx.x * blockDim.x + threadIdx.x;
  if (e >= E) return;
  atomicAdd(deg + dst[e], 1.f);
}

// ---------------- fused NNConv message kernel ----------------
// block = 256 (8 waves) handles 16 edges. Each wave: 8 WMMA col-tiles of the
// 16x1024 Wedge tile -> LDS (bf16), then 2 edges of matvec + atomic scatter.
__global__ void k_msg(const __bf16* __restrict__ h1, const __bf16* __restrict__ pb,
                      const float* __restrict__ nn2b, const float* __restrict__ h,
                      const int* __restrict__ src, const int* __restrict__ dst,
                      float* __restrict__ agg) {
  __shared__ __bf16 wt[16 * 1024];
  int lane = threadIdx.x & 31, wid = threadIdx.x >> 5;
  int tile = blockIdx.x;

  // A fragment: 16 edges x K=32 of h1 (shared across this wave's col tiles)
  int erow = tile * 16 + (lane & 15);
  int kb = (lane >> 4) * 8;
  AFrag a;
  a.q[0] = *(const uint4*)(h1 + erow * 32 + kb);
  a.q[1] = *(const uint4*)(h1 + erow * 32 + kb + 16);

#pragma unroll
  for (int t = 0; t < 8; ++t) {
    int ct = wid * 8 + t;
    AFrag bf;
    const uint4* bp = (const uint4*)(pb + (ct * 32 + lane) * 16);
    bf.q[0] = bp[0]; bf.q[1] = bp[1];
    float bias = nn2b[ct * 16 + (lane & 15)];
    v8f c;
#pragma unroll
    for (int r = 0; r < 8; ++r) c[r] = bias;
    c = __builtin_amdgcn_wmma_f32_16x16x32_bf16(false, a.v, false, bf.v,
                                                (short)0, c, false, false);
    int n0 = ct * 16 + (lane & 15);
    int mb = (lane >> 4) * 8;
#pragma unroll
    for (int r = 0; r < 8; ++r) wt[(mb + r) * 1024 + n0] = (__bf16)c[r];
  }
  __syncthreads();

  // consume: msg[e][o] = sum_i h[src[e]][i] * Wedge[e][i*32+o]
  for (int eL = wid; eL < 16; eL += 8) {
    int e = tile * 16 + eL;
    int s = src[e], dn = dst[e];
    float sv = h[s * 32 + lane];
    float acc = 0.f;
#pragma unroll
    for (int i = 0; i < 32; ++i)
      acc += __shfl(sv, i, 32) * (float)wt[eL * 1024 + i * 32 + lane];
    atomicAdd(agg + dn * 32 + lane, acc);
  }
}

// ---------------- conv-root + GRU step (wave per node, lane = feature) ----------------
__global__ void k_node(float* __restrict__ h, const float* __restrict__ agg,
                       const float* __restrict__ deg, const float* __restrict__ wr,
                       const float* __restrict__ cb, const float* __restrict__ wih,
                       const float* __restrict__ whh, const float* __restrict__ bih,
                       const float* __restrict__ bhh) {
  int lane = threadIdx.x & 31, wid = threadIdx.x >> 5;
  int n = blockIdx.x * 8 + wid;
  if (n >= N) return;
  float hv = h[n * 32 + lane];
  float acc = cb[lane];
#pragma unroll
  for (int i = 0; i < 32; ++i) acc += __shfl(hv, i, 32) * wr[i * 32 + lane];
  float m = acc + agg[n * 32 + lane] / fmaxf(deg[n], 1.f);
  m = fmaxf(m, 0.f);
  float gx0 = bih[lane], gx1 = bih[32 + lane], gx2 = bih[64 + lane];
  float gh0 = bhh[lane], gh1 = bhh[32 + lane], gh2 = bhh[64 + lane];
#pragma unroll
  for (int i = 0; i < 32; ++i) {
    float mi = __shfl(m, i, 32);
    float hi = __shfl(hv, i, 32);
    gx0 += mi * wih[i * 96 + lane];
    gx1 += mi * wih[i * 96 + 32 + lane];
    gx2 += mi * wih[i * 96 + 64 + lane];
    gh0 += hi * whh[i * 96 + lane];
    gh1 += hi * whh[i * 96 + 32 + lane];
    gh2 += hi * whh[i * 96 + 64 + lane];
  }
  float r = 1.f / (1.f + __expf(-(gx0 + gh0)));
  float z = 1.f / (1.f + __expf(-(gx1 + gh1)));
  float nn = tanhf(gx2 + r * gh2);
  h[n * 32 + lane] = (1.f - z) * nn + z * hv;
}

// ---------------- Set2Set ----------------
__global__ void k_lstm(const float* __restrict__ wih, const float* __restrict__ whh,
                       const float* __restrict__ bih, const float* __restrict__ bhh,
                       float* __restrict__ hl, float* __restrict__ cl, float* __restrict__ q,
                       float* __restrict__ qstar, float* __restrict__ rvec,
                       float* __restrict__ sumexp) {
  __shared__ float g[128];
  __shared__ float qs[64], hls[32];
  int tid = threadIdx.x;
  if (tid < 64) qs[tid] = qstar[tid];
  if (tid < 32) hls[tid] = hl[tid];
  __syncthreads();
  float acc = bih[tid] + bhh[tid];
  for (int k = 0; k < 64; ++k) acc += qs[k] * wih[k * 128 + tid];
  for (int k = 0; k < 32; ++k) acc += hls[k] * whh[k * 128 + tid];
  g[tid] = acc;
  __syncthreads();
  if (tid < 32) {
    float i = 1.f / (1.f + __expf(-g[tid]));
    float f = 1.f / (1.f + __expf(-g[32 + tid]));
    float gg = tanhf(g[64 + tid]);
    float o = 1.f / (1.f + __expf(-g[96 + tid]));
    float c = f * cl[tid] + i * gg;
    cl[tid] = c;
    float hh = o * tanhf(c);
    hl[tid] = hh;
    q[tid] = hh;
    rvec[tid] = 0.f;  // prep for attention pass
  }
  if (tid == 127) *sumexp = 0.f;
}

__global__ void k_att1(const float* __restrict__ h, const float* __restrict__ q,
                       float* __restrict__ expv, float* __restrict__ sumexp) {
  __shared__ float bsum;
  int lane = threadIdx.x & 31, wid = threadIdx.x >> 5;
  if (threadIdx.x == 0) bsum = 0.f;
  __syncthreads();
  int n = blockIdx.x * 8 + wid;
  if (n < N) {
    float d = h[n * 32 + lane] * q[lane];
#pragma unroll
    for (int o = 16; o; o >>= 1) d += __shfl_xor(d, o, 32);
    if (lane == 0) {
      float e = __expf(d);
      expv[n] = e;
      atomicAdd(&bsum, e);
    }
  }
  __syncthreads();
  if (threadIdx.x == 0) atomicAdd(sumexp, bsum);
}

__global__ void k_att2(const float* __restrict__ h, const float* __restrict__ expv,
                       float* __restrict__ rvec) {
  __shared__ float bacc[32];
  int lane = threadIdx.x & 31, wid = threadIdx.x >> 5;
  if (threadIdx.x < 32) bacc[threadIdx.x] = 0.f;
  __syncthreads();
  float acc = 0.f;
  for (int n = blockIdx.x * 8 + wid; n < N; n += gridDim.x * 8)
    acc += expv[n] * h[n * 32 + lane];
  atomicAdd(&bacc[lane], acc);
  __syncthreads();
  if (threadIdx.x < 32) atomicAdd(rvec + threadIdx.x, bacc[threadIdx.x]);
}

__global__ void k_qstar(const float* __restrict__ q, const float* __restrict__ rvec,
                        const float* __restrict__ sumexp, float* __restrict__ qstar) {
  int t = threadIdx.x;
  qstar[t] = (t < 32) ? q[t] : rvec[t - 32] / (*sumexp);
}

// ---------------- head ----------------
// cat[t][c]: faithful view/permute semantics from the reference
__global__ void k_cat(const float* __restrict__ h, const int* __restrict__ nonring,
                      const float* __restrict__ qstar, __bf16* __restrict__ cat) {
  int i = blockIdx.x * blockDim.x + threadIdx.x;
  if (i >= T * 192) return;
  int t = i / 192, c = i % 192;
  float v;
  if (c < 128) {
    int row = c * 256 + (t >> 5);        // a*T/D + t/D  (T=8192, D=32)
    int node = nonring[row];
    v = h[node * 32 + (t & 31)];
  } else {
    int j = c - 128;
    v = qstar[(t * 64 + j) >> 13];       // repeat(pool, T).reshape(T, 64)
  }
  cat[t * 192 + c] = (__bf16)v;
}

// block 64 = 2 waves; wave nt owns output col-tile nt; fused lin1 (WMMA) + relu + lin2
__global__ void k_head(const __bf16* __restrict__ cat, const __bf16* __restrict__ pb1,
                       const float* __restrict__ b1, const float* __restrict__ w2,
                       const float* __restrict__ b2, float* __restrict__ out) {
  __shared__ float yt[16][32];
  int lane = threadIdx.x & 31, nt = threadIdx.x >> 5;
  int tile = blockIdx.x;
  int erow = tile * 16 + (lane & 15);
  float bias = b1[nt * 16 + (lane & 15)];
  v8f c;
#pragma unroll
  for (int r = 0; r < 8; ++r) c[r] = bias;
#pragma unroll
  for (int kt = 0; kt < 6; ++kt) {
    AFrag a;
    int base = kt * 32 + (lane >> 4) * 8;
    a.q[0] = *(const uint4*)(cat + erow * 192 + base);
    a.q[1] = *(const uint4*)(cat + erow * 192 + base + 16);
    AFrag b;
    const uint4* bp = (const uint4*)(pb1 + ((kt * 2 + nt) * 32 + lane) * 16);
    b.q[0] = bp[0]; b.q[1] = bp[1];
    c = __builtin_amdgcn_wmma_f32_16x16x32_bf16(false, a.v, false, b.v,
                                                (short)0, c, false, false);
  }
  int n0 = nt * 16 + (lane & 15);
  int mb = (lane >> 4) * 8;
#pragma unroll
  for (int r = 0; r < 8; ++r) yt[mb + r][n0] = fmaxf(c[r], 0.f);
  __syncthreads();
  for (int idx = threadIdx.x; idx < 96; idx += 64) {
    int m = idx / 6, aa = idx % 6;
    float acc = b2[aa];
#pragma unroll
    for (int d = 0; d < 32; ++d) acc += yt[m][d] * w2[d * 6 + aa];
    out[(tile * 16 + m) * 6 + aa] = acc;
  }
}

// ---------------- launcher ----------------
extern "C" void kernel_launch(void* const* d_in, const int* in_sizes, int n_in,
                              void* d_out, int out_size, void* d_ws, size_t ws_size,
                              hipStream_t stream) {
  const float* x        = (const float*)d_in[0];
  const int*   eidx     = (const int*)d_in[1];
  const float* eattr    = (const float*)d_in[2];
  const int*   nonring  = (const int*)d_in[3];
  const float* lin0_w   = (const float*)d_in[4];
  const float* lin0_b   = (const float*)d_in[5];
  const float* nn1_w    = (const float*)d_in[6];
  const float* nn1_b    = (const float*)d_in[7];
  const float* nn2_w    = (const float*)d_in[8];
  const float* nn2_b    = (const float*)d_in[9];
  const float* conv_root= (const float*)d_in[10];
  const float* conv_b   = (const float*)d_in[11];
  const float* gru_w_ih = (const float*)d_in[12];
  const float* gru_w_hh = (const float*)d_in[13];
  const float* gru_b_ih = (const float*)d_in[14];
  const float* gru_b_hh = (const float*)d_in[15];
  const float* lstm_w_ih= (const float*)d_in[16];
  const float* lstm_w_hh= (const float*)d_in[17];
  const float* lstm_b_ih= (const float*)d_in[18];
  const float* lstm_b_hh= (const float*)d_in[19];
  const float* lin1_w   = (const float*)d_in[20];
  const float* lin1_b   = (const float*)d_in[21];
  const float* lin2_w   = (const float*)d_in[22];
  const float* lin2_b   = (const float*)d_in[23];

  const int* src = eidx;
  const int* dst = eidx + E;

  char* ws = (char*)d_ws;
  size_t off = 0;
  auto take = [&](size_t bytes) { char* p = ws + off; off += (bytes + 255) & ~(size_t)255; return p; };
  float*  h     = (float*)take((size_t)N * D * 4);
  float*  agg   = (float*)take((size_t)N * D * 4);
  float*  deg   = (float*)take((size_t)N * 4);
  float*  expv  = (float*)take((size_t)N * 4);
  __bf16* h1    = (__bf16*)take((size_t)E * D * 2);
  __bf16* pbnn2 = (__bf16*)take((size_t)64 * 32 * 16 * 2);
  __bf16* pbl1  = (__bf16*)take((size_t)6 * 2 * 32 * 16 * 2);
  __bf16* cat   = (__bf16*)take((size_t)T * 192 * 2);
  float*  s2s   = (float*)take(256 * 4);
  float* hl = s2s, *cl = s2s + 32, *q = s2s + 64, *rvec = s2s + 96;
  float* qstar = s2s + 128, *sumexp = s2s + 192;

  hipMemsetAsync(deg, 0, (size_t)N * 4, stream);
  hipMemsetAsync(s2s, 0, 256 * 4, stream);

  k_node_init<<<(N * D + 255) / 256, 256, 0, stream>>>(x, lin0_w, lin0_b, h);
  k_edge_h1<<<(E * D + 255) / 256, 256, 0, stream>>>(eattr, nn1_w, nn1_b, h1);
  k_pack_nn2<<<(64 * 32 * 16) / 256, 256, 0, stream>>>(nn2_w, pbnn2);
  k_pack_lin1<<<(6 * 2 * 32 * 16 + 255) / 256, 256, 0, stream>>>(lin1_w, pbl1);
  k_deg<<<(E + 255) / 256, 256, 0, stream>>>(dst, deg);

  for (int it = 0; it < 6; ++it) {
    hipMemsetAsync(agg, 0, (size_t)N * D * 4, stream);
    k_msg<<<E / 16, 256, 0, stream>>>(h1, pbnn2, nn2_b, h, src, dst, agg);
    k_node<<<(N + 7) / 8, 256, 0, stream>>>(h, agg, deg, conv_root, conv_b,
                                            gru_w_ih, gru_w_hh, gru_b_ih, gru_b_hh);
  }

  for (int it = 0; it < 6; ++it) {
    k_lstm<<<1, 128, 0, stream>>>(lstm_w_ih, lstm_w_hh, lstm_b_ih, lstm_b_hh,
                                  hl, cl, q, qstar, rvec, sumexp);
    k_att1<<<(N + 7) / 8, 256, 0, stream>>>(h, q, expv, sumexp);
    k_att2<<<256, 256, 0, stream>>>(h, expv, rvec);
    k_qstar<<<1, 64, 0, stream>>>(q, rvec, sumexp, qstar);
  }

  k_cat<<<(T * 192 + 255) / 256, 256, 0, stream>>>(h, nonring, qstar, cat);
  k_head<<<T / 16, 64, 0, stream>>>(cat, pbl1, lin1_b, lin2_w, lin2_b, (float*)d_out);
}
